// NonLocalBlock_71373766525083
// MI455X (gfx1250) — matrix operand order repaired and checked by reading, not compiled, not run on hardware
//
#include <hip/hip_runtime.h>

typedef __attribute__((ext_vector_type(16))) __bf16 v16bf;
typedef __attribute__((ext_vector_type(8)))  __bf16 v8bf;
typedef __attribute__((ext_vector_type(8)))  float  v8f;
typedef __attribute__((ext_vector_type(4)))  int    v4i_t;

#define HWD 4096
#define CI  128
#define CC  256
#define NB  4
#define QW  8          // waves (query tiles) per attention block

// workspace element offsets (bf16 elements)
static constexpr size_t OFF_WTH = 0;                          // theta_w  [128,256]
static constexpr size_t OFF_WPH = 32768;                      // phi_w    [128,256]
static constexpr size_t OFF_WG  = 65536;                      // g_w      [128,256]
static constexpr size_t OFF_WZ  = 98304;                      // wz_w     [256,128]
static constexpr size_t OFF_TH  = 131072;                     // thetaT   [B,HW,Ci]
static constexpr size_t OFF_PH  = OFF_TH + (size_t)NB*HWD*CI; // phiT     [B,HW,Ci]
static constexpr size_t OFF_GC  = OFF_PH + (size_t)NB*HWD*CI; // g        [B,Ci,HW]
static constexpr size_t OFF_Y   = OFF_GC + (size_t)NB*HWD*CI; // y        [B,HW,Ci]

// ---- async global->LDS (ASYNCcnt) with safe fallback --------------------------
#if defined(__HIP_DEVICE_COMPILE__) && __has_builtin(__builtin_amdgcn_global_load_async_to_lds_b128)
#define HAS_ASYNC 1
#else
#define HAS_ASYNC 0
#endif

__device__ __forceinline__ void copy16(const __bf16* g, __bf16* l)
{
#if HAS_ASYNC
    __builtin_amdgcn_global_load_async_to_lds_b128(
        (__attribute__((address_space(1))) v4i_t*)g,
        (__attribute__((address_space(3))) v4i_t*)l, 0, 0);
#else
    *(v8bf*)l = *(const v8bf*)g;
#endif
}

#if HAS_ASYNC
#if defined(__HIP_DEVICE_COMPILE__) && __has_builtin(__builtin_amdgcn_s_wait_asynccnt)
#define WAIT_ASYNC(N) __builtin_amdgcn_s_wait_asynccnt(N)
#else
#define WAIT_ASYNC(N) asm volatile("s_wait_asynccnt %0" :: "i"(N))
#endif
#else
#define WAIT_ASYNC(N)
#endif

// ---------------------------------------------------------------- weights->bf16
__global__ __launch_bounds__(256)
void wconv_kernel(const float* __restrict__ th, const float* __restrict__ ph,
                  const float* __restrict__ g,  const float* __restrict__ wz,
                  __bf16* __restrict__ ws)
{
    int i = blockIdx.x * blockDim.x + threadIdx.x;
    if (i >= 4 * 32768) return;
    int m = i >> 15, r = i & 32767;
    const float* src = (m == 0) ? th : (m == 1) ? ph : (m == 2) ? g : wz;
    ws[(size_t)m * 32768 + r] = (__bf16)src[r];
}

// ---------------------------------------------------------------- projections
// 8 waves / block, each wave owns one 16-col hw tile (128 spatial cols per block,
// staged once into LDS) and computes all 24 output tiles -> 8x less weight traffic.
// theta/phi stored transposed [HW,Ci]; g stored [Ci,HW].
__global__ __launch_bounds__(256)
void proj_kernel(const float* __restrict__ x,
                 const float* __restrict__ g_b, const float* __restrict__ th_b,
                 const float* __restrict__ ph_b, __bf16* __restrict__ ws)
{
    __shared__ __bf16 xs[8][16][CC + 8];          // 8 hw tiles, transposed bf16
    const int b   = blockIdx.x >> 5;              // 32 blocks per batch
    const int hw0 = (blockIdx.x & 31) << 7;       // 128 spatial cols per block
    const int t   = threadIdx.x;

    const float* xb = x + (size_t)b * CC * HWD;
    for (int e = t; e < 128 * CC; e += 256) {
        const int c = e >> 7, i = e & 127;        // coalesced 128-float rows
        xs[i >> 4][i & 15][c] = (__bf16)xb[(size_t)c * HWD + hw0 + i];
    }
    __syncthreads();

    const int wv = t >> 5, lane = t & 31;
    const int n = lane & 15, half = lane >> 4, ka = half * 8;
    const int hwt = hw0 + (wv << 4);              // this wave's hw tile

    #pragma unroll
    for (int p = 0; p < 3; ++p) {
        const __bf16* W = (p == 0) ? (ws + OFF_WTH) : (p == 1) ? (ws + OFF_WPH)
                                                               : (ws + OFF_WG);
        const float* bias = (p == 0) ? th_b : (p == 1) ? ph_b : g_b;
        for (int mt = 0; mt < 8; ++mt) {
            const int ci0 = mt << 4;
            const __bf16* wr = W + (size_t)(ci0 + n) * CC;

            v16bf a[8];
            #pragma unroll
            for (int kc = 0; kc < 8; ++kc) {
                const int k0 = kc << 5;
                *(v8bf*)&a[kc]       = *(const v8bf*)(wr + k0 + ka);
                *((v8bf*)&a[kc] + 1) = *(const v8bf*)(wr + k0 + 16 + ka);
            }
            v8f acc = {};
            #pragma unroll
            for (int kc = 0; kc < 8; ++kc) {
                const int k0 = kc << 5;
                v16bf bm;
                *(v8bf*)&bm       = *(const v8bf*)(&xs[wv][n][k0 + ka]);
                *((v8bf*)&bm + 1) = *(const v8bf*)(&xs[wv][n][k0 + 16 + ka]);
                acc = __builtin_amdgcn_wmma_f32_16x16x32_bf16(false, a[kc], false, bm,
                                                              (short)0, acc, false, false);
            }
            if (p < 2) {
                __bf16* dst = ws + ((p == 0) ? OFF_TH : OFF_PH) + (size_t)b * HWD * CI;
                #pragma unroll
                for (int i = 0; i < 8; ++i) {
                    const int row = ci0 + i + half * 8;
                    dst[(size_t)(hwt + n) * CI + row] = (__bf16)(acc[i] + bias[row]);
                }
            } else {
                __bf16* dst = ws + OFF_GC + (size_t)b * CI * HWD;
                #pragma unroll
                for (int i = 0; i < 8; ++i) {
                    const int row = ci0 + i + half * 8;
                    dst[(size_t)row * HWD + hwt + n] = (__bf16)(acc[i] + bias[row]);
                }
            }
        }
    }
}

// ---------------------------------------------------------------- flash attention
// 8 waves / block, one 16-row query tile each, sharing a double-buffered
// K/V stream staged in LDS via async global->LDS copies.
__global__ __launch_bounds__(32 * QW)
void attn_kernel(__bf16* __restrict__ ws)
{
    __shared__ __bf16 kls[2][32][CI + 8];     // K tile  [32 keys][128 ch]
    __shared__ __bf16 vls[2][CI][32 + 16];    // V tile  [128 ch][32 keys]
    __shared__ __bf16 plds[QW][16][40];       // per-wave P repack

    const int b    = blockIdx.x >> 5;
    const int t    = threadIdx.x;
    const int wv   = t >> 5;
    const int lane = t & 31;
    const int n = lane & 15, half = lane >> 4, ka = half * 8;
    const int q0 = ((blockIdx.x & 31) * QW + wv) << 4;

    const __bf16* TH = ws + OFF_TH + (size_t)b * HWD * CI;
    const __bf16* PH = ws + OFF_PH + (size_t)b * HWD * CI;
    const __bf16* GC = ws + OFF_GC + (size_t)b * CI * HWD;
    __bf16* Y        = ws + OFF_Y  + (size_t)b * HWD * CI;

    // cooperative stage of one 32-key K/V tile: 512 + 512 16-byte chunks
    auto stage = [&](int buf, int j0) {
        #pragma unroll
        for (int s = 0; s < 2; ++s) {
            const int c = t + s * 256;
            const int kr = c >> 4, kcol = c & 15;           // K: 32 rows x 16 chunks
            copy16(PH + (size_t)(j0 + kr) * CI + kcol * 8, &kls[buf][kr][kcol * 8]);
            const int vr = c >> 2, vcol = c & 3;            // V: 128 rows x 4 chunks
            copy16(GC + (size_t)vr * HWD + j0 + vcol * 8, &vls[buf][vr][vcol * 8]);
        }
    };

    // Q tile [16 x 128] in A layout
    v16bf qa[4];
    const __bf16* qrow = TH + (size_t)(q0 + n) * CI;
    #pragma unroll
    for (int kc = 0; kc < 4; ++kc) {
        const int k0 = kc * 32;
        *(v8bf*)&qa[kc]       = *(const v8bf*)(qrow + k0 + ka);
        *((v8bf*)&qa[kc] + 1) = *(const v8bf*)(qrow + k0 + 16 + ka);
    }

    v8f o[8];
    float m[8], l[8];
    const v8f vzero = {};
    #pragma unroll
    for (int i = 0; i < 8; ++i) { o[i] = vzero; m[i] = -1e30f; l[i] = 0.f; }

    stage(0, 0);
    WAIT_ASYNC(0);
    __syncthreads();

    for (int j0 = 0; j0 < HWD; j0 += 32) {
        const int cur = (j0 >> 5) & 1;
        const bool nxt = (j0 + 32) < HWD;
        if (nxt) {                       // issue next tile, overlap with compute
            stage(cur ^ 1, j0 + 32);
            WAIT_ASYNC(4);               // current tile's copies done; next in flight
        } else {
            WAIT_ASYNC(0);
        }
        __syncthreads();                 // current tile visible to all waves

        // ---- S = Q K^T (two 16-key tiles), operands preloaded from LDS
        v16bf kb[8];
        #pragma unroll
        for (int kc = 0; kc < 4; ++kc) {
            const int k0 = kc * 32 + ka;
            *(v8bf*)&kb[kc]           = *(const v8bf*)(&kls[cur][n][k0]);
            *((v8bf*)&kb[kc] + 1)     = *(const v8bf*)(&kls[cur][n][k0 + 16]);
            *(v8bf*)&kb[4 + kc]       = *(const v8bf*)(&kls[cur][16 + n][k0]);
            *((v8bf*)&kb[4 + kc] + 1) = *(const v8bf*)(&kls[cur][16 + n][k0 + 16]);
        }
        v8f s0 = {}, s1 = {};
        #pragma unroll
        for (int kc = 0; kc < 4; ++kc) {
            s0 = __builtin_amdgcn_wmma_f32_16x16x32_bf16(false, qa[kc], false, kb[kc],
                                                         (short)0, s0, false, false);
            s1 = __builtin_amdgcn_wmma_f32_16x16x32_bf16(false, qa[kc], false, kb[4 + kc],
                                                         (short)0, s1, false, false);
        }

        // ---- online softmax (row stats replicated across each 16-lane half)
        float sc[8];
        #pragma unroll
        for (int i = 0; i < 8; ++i) {
            float tmx = fmaxf(s0[i], s1[i]);
            tmx = fmaxf(tmx, __shfl_xor(tmx, 1, 32));
            tmx = fmaxf(tmx, __shfl_xor(tmx, 2, 32));
            tmx = fmaxf(tmx, __shfl_xor(tmx, 4, 32));
            tmx = fmaxf(tmx, __shfl_xor(tmx, 8, 32));
            const float mn = fmaxf(m[i], tmx);
            sc[i] = __expf(m[i] - mn);
            m[i] = mn;
            const float p0 = __expf(s0[i] - mn);
            const float p1 = __expf(s1[i] - mn);
            s0[i] = p0; s1[i] = p1;
            float su = p0 + p1;
            su += __shfl_xor(su, 1, 32);
            su += __shfl_xor(su, 2, 32);
            su += __shfl_xor(su, 4, 32);
            su += __shfl_xor(su, 8, 32);
            l[i] = l[i] * sc[i] + su;
        }
        #pragma unroll
        for (int nt = 0; nt < 8; ++nt) {
            #pragma unroll
            for (int i = 0; i < 8; ++i) o[nt][i] *= sc[i];
        }

        // ---- P (C-layout f32) -> A-layout bf16 via per-wave LDS patch
        #pragma unroll
        for (int i = 0; i < 8; ++i) {
            const int row = i + half * 8;
            plds[wv][row][n]      = (__bf16)s0[i];
            plds[wv][row][16 + n] = (__bf16)s1[i];
        }
        v16bf pa;
        *(v8bf*)&pa       = *(const v8bf*)(&plds[wv][n][ka]);
        *((v8bf*)&pa + 1) = *(const v8bf*)(&plds[wv][n][16 + ka]);

        // ---- O += P V, V-tile operands preloaded from LDS
        v16bf gb[8];
        #pragma unroll
        for (int nt = 0; nt < 8; ++nt) {
            const __bf16* gr = &vls[cur][nt * 16 + n][ka];
            *(v8bf*)&gb[nt]       = *(const v8bf*)gr;
            *((v8bf*)&gb[nt] + 1) = *(const v8bf*)(gr + 16);
        }
        #pragma unroll
        for (int nt = 0; nt < 8; ++nt)
            o[nt] = __builtin_amdgcn_wmma_f32_16x16x32_bf16(false, pa, false, gb[nt],
                                                            (short)0, o[nt], false, false);

        __syncthreads();                 // all waves done reading `cur` buffer
    }

    // ---- normalize and store y [HW,Ci] bf16
    #pragma unroll
    for (int i = 0; i < 8; ++i) l[i] = 1.0f / l[i];
    #pragma unroll
    for (int nt = 0; nt < 8; ++nt) {
        #pragma unroll
        for (int i = 0; i < 8; ++i) {
            const int row = q0 + i + half * 8;
            Y[(size_t)row * CI + nt * 16 + n] = (__bf16)(o[nt][i] * l[i]);
        }
    }
}

// ---------------------------------------------------------------- z = Wz y + b + x
// 8 waves / block, each wave owns one hw tile; Wz staged once per block into LDS
// via async copies.
__global__ __launch_bounds__(256)
void zout_kernel(const float* __restrict__ x, const float* __restrict__ wz_b,
                 const __bf16* __restrict__ ws, float* __restrict__ out)
{
    __shared__ __bf16 wzl[CC][CI + 8];
    const int b   = blockIdx.x >> 5;
    const int hw0 = (blockIdx.x & 31) << 7;
    const int t = threadIdx.x, wv = t >> 5, lane = t & 31;
    const int n = lane & 15, half = lane >> 4, ka = half * 8;

    const __bf16* WZ = ws + OFF_WZ;
    for (int c = t; c < CC * 16; c += 256) {          // 4096 chunks of 16 B
        const int row = c >> 4, col = c & 15;
        copy16(WZ + (size_t)row * CI + col * 8, &wzl[row][col * 8]);
    }
    WAIT_ASYNC(0);
    __syncthreads();

    const int hwt = hw0 + (wv << 4);
    const __bf16* Y = ws + OFF_Y + (size_t)b * HWD * CI;
    const float* xb = x + (size_t)b * CC * HWD;
    float* ob       = out + (size_t)b * CC * HWD;

    const __bf16* yr = Y + (size_t)(hwt + n) * CI;
    v16bf bm[4];
    #pragma unroll
    for (int kc = 0; kc < 4; ++kc) {
        const int k0 = kc * 32 + ka;
        *(v8bf*)&bm[kc]       = *(const v8bf*)(yr + k0);
        *((v8bf*)&bm[kc] + 1) = *(const v8bf*)(yr + k0 + 16);
    }

    for (int mt = 0; mt < 16; ++mt) {
        const int o0 = mt << 4;
        v16bf a[4];
        #pragma unroll
        for (int kc = 0; kc < 4; ++kc) {
            const int k0 = kc * 32;
            *(v8bf*)&a[kc]       = *(const v8bf*)(&wzl[o0 + n][k0 + ka]);
            *((v8bf*)&a[kc] + 1) = *(const v8bf*)(&wzl[o0 + n][k0 + 16 + ka]);
        }
        v8f acc = {};
        #pragma unroll
        for (int kc = 0; kc < 4; ++kc)
            acc = __builtin_amdgcn_wmma_f32_16x16x32_bf16(false, a[kc], false, bm[kc],
                                                          (short)0, acc, false, false);
        #pragma unroll
        for (int i = 0; i < 8; ++i) {
            const int row = o0 + i + half * 8;
            const size_t idx = (size_t)row * HWD + hwt + n;
            ob[idx] = acc[i] + wz_b[row] + xb[idx];
        }
    }
}

extern "C" void kernel_launch(void* const* d_in, const int* in_sizes, int n_in,
                              void* d_out, int out_size, void* d_ws, size_t ws_size,
                              hipStream_t stream)
{
    (void)in_sizes; (void)n_in; (void)out_size; (void)ws_size;
    const float* x    = (const float*)d_in[0];
    const float* g_w  = (const float*)d_in[1];
    const float* g_b  = (const float*)d_in[2];
    const float* th_w = (const float*)d_in[3];
    const float* th_b = (const float*)d_in[4];
    const float* ph_w = (const float*)d_in[5];
    const float* ph_b = (const float*)d_in[6];
    const float* wz_w = (const float*)d_in[7];
    const float* wz_b = (const float*)d_in[8];
    __bf16* ws = (__bf16*)d_ws;
    float* out = (float*)d_out;

    wconv_kernel<<<512, 256, 0, stream>>>(th_w, ph_w, g_w, wz_w, ws);
    proj_kernel <<<NB * (HWD / 128), 256, 0, stream>>>(x, g_b, th_b, ph_b, ws);
    attn_kernel <<<NB * (HWD / (16 * QW)), 32 * QW, 0, stream>>>(ws);
    zout_kernel <<<NB * (HWD / 128), 256, 0, stream>>>(x, wz_b, ws, out);
}
